// SupervisedContrastiveLoss_86681029967953
// MI455X (gfx1250) — matrix operand order repaired and checked by reading, not compiled, not run on hardware
//
#include <hip/hip_runtime.h>
#include <hip/hip_bf16.h>
#include <math.h>

// ---------------------------------------------------------------------------
// Supervised contrastive loss w/ hard-negative mining, MI455X (gfx1250)
//   N=4096 samples, D=1024 dims, labels in [0,256)
//   sim = normalize(f) @ normalize(f)^T / 0.07 computed with
//   v_wmma_f32_16x16x32_f16; per-row k-th-largest-negative threshold found by
//   4-pass MSB radix select on LDS histograms; loss reduced with f32 atomics.
// ---------------------------------------------------------------------------

typedef _Float16 v8h  __attribute__((ext_vector_type(8)));
typedef _Float16 v16h __attribute__((ext_vector_type(16)));
typedef float    v8f  __attribute__((ext_vector_type(8)));

constexpr int   NROW = 4096;
constexpr int   NDIM = 1024;
constexpr float INV_T = 1.0f / 0.07f;   // 1 / TEMPERATURE
constexpr int   BK   = 32;              // K tile (one WMMA K per step)
constexpr int   LDH  = 40;              // padded LDS row stride (halves): 80B, keeps 16B alignment

// monotone map f32 -> u32 (larger float => larger key); no NaNs occur here
__device__ __forceinline__ unsigned f2key(float x) {
  unsigned u = __float_as_uint(x);
  return (u & 0x80000000u) ? ~u : (u | 0x80000000u);
}

// ---------------------------------------------------------------------------
// Kernel 1: L2-normalize each row (torch-style x / max(||x||, 1e-12)),
// emit f16 copy for the WMMA GEMM.  One 256-thread block per row.
// ---------------------------------------------------------------------------
__global__ __launch_bounds__(256) void normalize_f16(
    const float* __restrict__ f, _Float16* __restrict__ fh) {
  const int row = blockIdx.x, tid = threadIdx.x;
  const float* fr = f + (size_t)row * NDIM;

  float v[4];
  float ss = 0.0f;
#pragma unroll
  for (int t = 0; t < 4; ++t) {
    v[t] = fr[tid + 256 * t];
    ss += v[t] * v[t];
  }
  __shared__ float red[256];
  red[tid] = ss;
  __syncthreads();
  for (int s = 128; s > 0; s >>= 1) {
    if (tid < s) red[tid] += red[tid + s];
    __syncthreads();
  }
  const float scale = 1.0f / fmaxf(sqrtf(red[0]), 1e-12f);

  _Float16* orow = fh + (size_t)row * NDIM;
#pragma unroll
  for (int t = 0; t < 4; ++t) orow[tid + 256 * t] = (_Float16)(v[t] * scale);
}

// ---------------------------------------------------------------------------
// Kernel 2: sim = (fh @ fh^T) * (1/T), 4096x4096 f32 output.
// Block = 256 threads (8 waves) computing a 128x128 tile; BK=32.
// Wave (wm,wn) in a 4x2 grid owns a 32x64 sub-tile = 2x4 WMMA 16x16 tiles.
// Fragment layouts follow CDNA5 ISA 16-bit A/B tables.
// ---------------------------------------------------------------------------
__global__ __launch_bounds__(256) void gemm_sim(
    const _Float16* __restrict__ fh, float* __restrict__ sim) {
  __shared__ __align__(16) _Float16 At[128 * LDH];
  __shared__ __align__(16) _Float16 Bt[128 * LDH];

  const int tid  = threadIdx.x;
  const int wave = tid >> 5;
  const int lane = tid & 31;
  const int wm   = wave >> 1;       // 0..3 -> 32-row strip
  const int wn   = wave & 1;        // 0..1 -> 64-col strip
  const int half = lane >> 4;       // lane group (0: lanes 0-15, 1: lanes 16-31)
  const int l15  = lane & 15;

  const int rowBase = blockIdx.y * 128;
  const int colBase = blockIdx.x * 128;

  // cooperative global->LDS staging: thread t loads 16 halves (32B) of one row
  const int lrow = tid >> 1;                // 0..127
  const int lcol = (tid & 1) * 16;          // 0 or 16
  const _Float16* gA = fh + (size_t)(rowBase + lrow) * NDIM + lcol;
  const _Float16* gB = fh + (size_t)(colBase + lrow) * NDIM + lcol;

  v8f acc[2][4] = {};

  for (int kt = 0; kt < NDIM / BK; ++kt) {
    const int kb = kt * BK;
    const uint4 a0 = ((const uint4*)(gA + kb))[0];
    const uint4 a1 = ((const uint4*)(gA + kb))[1];
    const uint4 b0 = ((const uint4*)(gB + kb))[0];
    const uint4 b1 = ((const uint4*)(gB + kb))[1];
    if (kt + 1 < NDIM / BK) {                 // global_prefetch_b8 of next K tile
      __builtin_prefetch(gA + kb + BK, 0, 1);
      __builtin_prefetch(gB + kb + BK, 0, 1);
    }
    __syncthreads();                          // previous iter's LDS reads done
    *(uint4*)(&At[lrow * LDH + lcol])     = a0;
    *(uint4*)(&At[lrow * LDH + lcol + 8]) = a1;
    *(uint4*)(&Bt[lrow * LDH + lcol])     = b0;
    *(uint4*)(&Bt[lrow * LDH + lcol + 8]) = b1;
    __syncthreads();

    // A fragment 16x32 f16: lane group 0 -> K 0-7 & 16-23, group 1 -> K 8-15 & 24-31
    v16h afr[2];
#pragma unroll
    for (int tm = 0; tm < 2; ++tm) {
      const _Float16* base = &At[(wm * 32 + tm * 16 + l15) * LDH + half * 8];
      v8h lo = *(const v8h*)base;
      v8h hi = *(const v8h*)(base + 16);
      afr[tm] = __builtin_shufflevector(lo, hi, 0, 1, 2, 3, 4, 5, 6, 7,
                                        8, 9, 10, 11, 12, 13, 14, 15);
    }
    // B fragment 32x16 (K x N): lane = column n, group 0 -> K 0-7 & 16-23, group 1 -> K 8-15 & 24-31
    v16h bfr[4];
#pragma unroll
    for (int tn = 0; tn < 4; ++tn) {
      const _Float16* base = &Bt[(wn * 64 + tn * 16 + l15) * LDH + half * 8];
      v8h lo = *(const v8h*)base;
      v8h hi = *(const v8h*)(base + 16);
      bfr[tn] = __builtin_shufflevector(lo, hi, 0, 1, 2, 3, 4, 5, 6, 7,
                                        8, 9, 10, 11, 12, 13, 14, 15);
    }
#pragma unroll
    for (int tm = 0; tm < 2; ++tm)
#pragma unroll
      for (int tn = 0; tn < 4; ++tn)
        acc[tm][tn] = __builtin_amdgcn_wmma_f32_16x16x32_f16(
            false, afr[tm], false, bfr[tn], (short)0, acc[tm][tn], false, false);
  }

  // C/D layout: VGPR r -> M = r + 8*halfgroup, N = lane&15
#pragma unroll
  for (int tm = 0; tm < 2; ++tm)
#pragma unroll
    for (int tn = 0; tn < 4; ++tn) {
      const int r0 = rowBase + wm * 32 + tm * 16 + half * 8;
      const int c0 = colBase + wn * 64 + tn * 16 + l15;
#pragma unroll
      for (int r = 0; r < 8; ++r)
        sim[(size_t)(r0 + r) * NROW + c0] = acc[tm][tn][r] * INV_T;
    }
}

// ---------------------------------------------------------------------------
// Kernel 3: per-row hard-negative threshold (exact k-th largest negative via
// 4-pass MSB radix select on LDS histograms) + loss contribution.
// One 256-thread block per row; row + byte-labels cached in LDS (~24 KB).
// ---------------------------------------------------------------------------
__global__ __launch_bounds__(256) void loss_rows(
    const float* __restrict__ sim, const int* __restrict__ labels,
    float* __restrict__ out) {
  __shared__ float         srow[NROW];
  __shared__ unsigned char lab8[NROW];
  __shared__ unsigned int  hist[256];
  __shared__ float         redf[256];
  __shared__ float         redg[256];
  __shared__ int           redi[256];
  __shared__ unsigned int  sh_prefix;
  __shared__ int           sh_k;

  const int i = blockIdx.x, tid = threadIdx.x;
  const unsigned char myLab = (unsigned char)labels[i];

  const float* rowp = sim + (size_t)i * NROW;
  int cnt = 0;
  for (int j = tid; j < NROW; j += 256) {
    const float s = rowp[j];
    srow[j] = s;
    const unsigned char lb = (unsigned char)labels[j];
    lab8[j] = lb;
    cnt += (lb != myLab);
  }
  redi[tid] = cnt;
  __syncthreads();
  for (int s = 128; s > 0; s >>= 1) {
    if (tid < s) redi[tid] += redi[tid + s];
    __syncthreads();
  }
  const int count = redi[0];
  if (tid == 0) {
    const int k = count >> 1;            // floor(count * 0.5)
    sh_k = (k < 1) ? 1 : k;              // max(1, ...)
    sh_prefix = 0u;
  }
  __syncthreads();

  // MSB-first radix select of the sh_k-th largest negative similarity
  for (int p = 0; p < 4; ++p) {
    hist[tid] = 0u;
    __syncthreads();
    const unsigned pref = sh_prefix;
    const int hishift = 32 - 8 * p;      // guarded at p==0 (short-circuit)
    const int bshift = 24 - 8 * p;
    for (int j = tid; j < NROW; j += 256) {
      if (lab8[j] != myLab) {
        const unsigned key = f2key(srow[j]);
        if (p == 0 || (key >> hishift) == pref)
          atomicAdd(&hist[(key >> bshift) & 255u], 1u);
      }
    }
    __syncthreads();
    if (tid == 0) {
      const unsigned kk = (unsigned)sh_k;
      unsigned cum = 0;
      for (int b = 255; b >= 0; --b) {
        const unsigned c = hist[b];
        if (cum + c >= kk) {
          sh_prefix = (pref << 8) | (unsigned)b;
          sh_k = (int)(kk - cum);        // rank within the chosen bucket
          break;
        }
        cum += c;
      }
    }
    __syncthreads();
  }
  const unsigned tkey = sh_prefix;       // exact key of threshold value

  // denominator & positive statistics (">= threshold" keeps ties, as reference)
  float negExp = 0.0f, posSum = 0.0f;
  int posCnt = 0;
  for (int j = tid; j < NROW; j += 256) {
    const float s = srow[j];
    if (lab8[j] != myLab) {
      if (f2key(s) >= tkey) negExp += expf(s);
    } else if (j != i) {
      posSum += s;
      posCnt++;
    }
  }
  redf[tid] = negExp;
  redg[tid] = posSum;
  redi[tid] = posCnt;
  __syncthreads();
  for (int s = 128; s > 0; s >>= 1) {
    if (tid < s) {
      redf[tid] += redf[tid + s];
      redg[tid] += redg[tid + s];
      redi[tid] += redi[tid + s];
    }
    __syncthreads();
  }
  if (tid == 0) {
    const float denom = redf[0] + expf(srow[i]) + 1e-10f;   // hardneg + self + eps
    const float pc = (float)redi[0];
    const float mlpp = (redg[0] - pc * logf(denom)) / (pc + 1e-10f);
    atomicAdd(out, -mlpp * (1.0f / (float)NROW));
  }
}

// ---------------------------------------------------------------------------
extern "C" void kernel_launch(void* const* d_in, const int* in_sizes, int n_in,
                              void* d_out, int out_size, void* d_ws, size_t ws_size,
                              hipStream_t stream) {
  (void)in_sizes; (void)n_in; (void)out_size; (void)ws_size;
  const float* features = (const float*)d_in[0];
  const int*   labels   = (const int*)d_in[1];
  float*       out      = (float*)d_out;

  _Float16* fh  = (_Float16*)d_ws;                                   // 8 MB
  float*    sim = (float*)((char*)d_ws + (size_t)NROW * NDIM * 2);   // 64 MB

  hipMemsetAsync(d_out, 0, sizeof(float), stream);
  normalize_f16<<<NROW, 256, 0, stream>>>(features, fh);
  gemm_sim<<<dim3(NROW / 128, NROW / 128), 256, 0, stream>>>(fh, sim);
  loss_rows<<<NROW, 256, 0, stream>>>(sim, labels, out);
}
